// LFA_55250459296229
// MI455X (gfx1250) — matrix operand order repaired and checked by reading, not compiled, not run on hardware
//
#include <hip/hip_runtime.h>
#include <math.h>

typedef _Float16 v16h __attribute__((ext_vector_type(16)));
typedef float    v8f  __attribute__((ext_vector_type(8)));
typedef float    v2f  __attribute__((ext_vector_type(2)));
typedef _Float16 h2   __attribute__((ext_vector_type(2)));

#define BATCH   4
#define NPTS    16384
#define KNN     16
#define DIM     128
#define D2      64
#define P_TOTAL (BATCH * NPTS)   /* 65536 points */
#define EPSV    1e-5f

/* ---------------- WMMA helpers ---------------- */

static __device__ __forceinline__ v8f wmma16(v16h a, v16h b, v8f c) {
  return __builtin_amdgcn_wmma_f32_16x16x32_f16(false, a, false, b, (short)0, c,
                                                false, false);
}

/* A-fragment (rows=M=16, K-tile of 32) from row-major f16 [16][stride] in LDS.
   Per ISA 16-bit A layout: lane(lo,hi): halfs 0..7  = K[k0 + hi*8 .. +7],
                                         halfs 8..15 = K[k0 + 16 + hi*8 .. +7]. */
static __device__ __forceinline__ v16h loadA(const _Float16* base, int stride,
                                             int k0, int lo, int hi) {
  const _Float16* r = base + lo * stride + k0 + hi * 8;
  v16h a;
#pragma unroll
  for (int j = 0; j < 8; ++j) { a[j] = r[j]; a[8 + j] = r[16 + j]; }
  return a;
}

/* A-fragment where all 16 rows are identical (broadcast row). */
static __device__ __forceinline__ v16h loadAb(const _Float16* row, int k0, int hi) {
  const _Float16* r = row + k0 + hi * 8;
  v16h a;
#pragma unroll
  for (int j = 0; j < 8; ++j) { a[j] = r[j]; a[8 + j] = r[16 + j]; }
  return a;
}

/* B-fragment for computing X @ W^T: B[k][n] = W[n][k], W row-major [OUT][IN] f16.
   Lane holds column n = t*16+lo, K = s*32 + hi*8 + 0..7 and s*32 + 16 + hi*8 + 0..7. */
static __device__ __forceinline__ v16h loadB(const _Float16* W, int IN, int t,
                                             int s, int lo, int hi) {
  const _Float16* r = W + (t * 16 + lo) * IN + s * 32 + hi * 8;
  v16h b;
#pragma unroll
  for (int j = 0; j < 8; ++j) { b[j] = r[j]; b[8 + j] = r[16 + j]; }
  return b;
}

/* ---------------- kernel 0: zero BN accumulators ---------------- */

__global__ void k_zero(float* __restrict__ acc) { acc[threadIdx.x] = 0.0f; }

/* ---------------- kernel 1: xf = x @ proj_w^T  (full fp32 WMMA) ----------------
   V_WMMA_F32_16X16X4_F32; A 16x4 f32: lane holds K = 2*hi + v. */
__global__ void __launch_bounds__(256) k_proj(const float* __restrict__ x,
                                              const float* __restrict__ pw,
                                              float* __restrict__ xf) {
  __shared__ float wS[DIM * DIM]; /* 64 KB */
  const int tid = threadIdx.x;
  for (int i = tid; i < DIM * DIM; i += 256) wS[i] = pw[i];
  __syncthreads();

  const int wave = tid >> 5, lane = tid & 31, lo = lane & 15, hi = lane >> 4;
  const int rt = blockIdx.x * 8 + wave; /* row tile: 16 rows */
  const float* arow = x + (size_t)(rt * 16 + lo) * DIM + 2 * hi;

  v8f acc[8];
#pragma unroll
  for (int t = 0; t < 8; ++t) acc[t] = (v8f){0,0,0,0,0,0,0,0};

#pragma unroll 4
  for (int k0 = 0; k0 < DIM; k0 += 4) {
    v2f a; a[0] = arow[k0]; a[1] = arow[k0 + 1];
#pragma unroll
    for (int t = 0; t < 8; ++t) {
      const float* br = &wS[(t * 16 + lo) * DIM + k0 + 2 * hi];
      v2f bb; bb[0] = br[0]; bb[1] = br[1];
      acc[t] = __builtin_amdgcn_wmma_f32_16x16x4_f32(false, a, false, bb,
                                                     (short)0, acc[t],
                                                     false, false);
    }
  }
#pragma unroll
  for (int t = 0; t < 8; ++t)
#pragma unroll
    for (int v = 0; v < 8; ++v)
      xf[(size_t)(rt * 16 + v + 8 * hi) * DIM + t * 16 + lo] = acc[t][v];
}

/* ---------------- kernel 2: fused LFA (gather + NCA MLPs + max + BN stats) ---- */

__global__ void __launch_bounds__(256) k_lfa(
    const float* __restrict__ xyz, const float* __restrict__ xf,
    const int* __restrict__ knn,
    const float* __restrict__ m1w, const float* __restrict__ m1b,
    const float* __restrict__ m2w, const float* __restrict__ m2b,
    const float* __restrict__ w1,  const float* __restrict__ b1,
    const float* __restrict__ w2,  const float* __restrict__ b2,
    float* __restrict__ out, float* __restrict__ gacc) {

  /* shared weights (f16) + biases (f32) */
  __shared__ __align__(16) _Float16 w1h[DIM * DIM];   /* 32 KB */
  __shared__ __align__(16) _Float16 w2h[DIM * DIM];   /* 32 KB */
  __shared__ __align__(16) _Float16 m2h[D2 * D2];     /*  8 KB */
  __shared__ float m1wS[D2 * 3];
  __shared__ float m1bS[D2], m2bS[D2], b1S[DIM], b2S[DIM];
  /* per-wave scratch (8 waves) */
  __shared__ __align__(16) _Float16 p1buf[8][16 * D2];   /* p0 then p1 */
  __shared__ __align__(16) _Float16 hbuf[8][16 * DIM];   /* gelu hidden */
  __shared__ __align__(16) _Float16 plocalS[8][D2];
  __shared__ float xyzS[8][16][3];
  __shared__ int   nbrS[8][16];
  /* BN statistic accumulators */
  __shared__ float sSum[DIM], sSumSq[DIM];

  const int tid = threadIdx.x;
  for (int i = tid; i < DIM * DIM; i += 256) {
    w1h[i] = (_Float16)w1[i];
    w2h[i] = (_Float16)w2[i];
  }
  for (int i = tid; i < D2 * D2; i += 256) m2h[i] = (_Float16)m2w[i];
  for (int i = tid; i < D2 * 3; i += 256)  m1wS[i] = m1w[i];
  if (tid < D2)  { m1bS[tid] = m1b[tid]; m2bS[tid] = m2b[tid]; }
  if (tid < DIM) { b1S[tid] = b1[tid]; b2S[tid] = b2[tid];
                   sSum[tid] = 0.0f; sSumSq[tid] = 0.0f; }
  __syncthreads();

  const int slot = tid >> 5;
  const int lane = tid & 31, lo = lane & 15, hi = lane >> 4;

  for (int it = 0; it < 8; ++it) {
    const int p = (blockIdx.x * 8 + slot) * 8 + it; /* 1024*8*8 == 65536 */
    const int b = p >> 14, n = p & (NPTS - 1);
    const int bOff = b << 14;

    /* ---- stage A0: neighbor indices + relative coords into LDS ---- */
    if (lane < 16) {
      const int nbr = knn[(size_t)p * KNN + lane];
      nbrS[slot][lane] = nbr;
      const float cx = xyz[(size_t)(bOff + n) * 3 + 0];
      const float cy = xyz[(size_t)(bOff + n) * 3 + 1];
      const float cz = xyz[(size_t)(bOff + n) * 3 + 2];
      const float* q = xyz + (size_t)(bOff + nbr) * 3;
      xyzS[slot][lane][0] = q[0] - cx;
      xyzS[slot][lane][1] = q[1] - cy;
      xyzS[slot][lane][2] = q[2] - cz;
    }
    asm volatile("s_wait_dscnt 0" ::: "memory");

    /* ---- stage A1: p0 = xyzs @ mlp1^T + b (tiny K=3, VALU); track p_local ---- */
    {
      const int d0 = lane << 1;
      const float* w0 = &m1wS[d0 * 3];
      const float w00 = w0[0], w01 = w0[1], w02 = w0[2];
      const float w10 = w0[3], w11 = w0[4], w12 = w0[5];
      const float bb0 = m1bS[d0], bb1 = m1bS[d0 + 1];
      float pm0 = -3.0e38f, pm1 = -3.0e38f;
#pragma unroll
      for (int k = 0; k < 16; ++k) {
        const float vx = xyzS[slot][k][0], vy = xyzS[slot][k][1],
                    vz = xyzS[slot][k][2];
        const float q0 = fmaf(w02, vz, fmaf(w01, vy, fmaf(w00, vx, bb0)));
        const float q1 = fmaf(w12, vz, fmaf(w11, vy, fmaf(w10, vx, bb1)));
        pm0 = fmaxf(pm0, q0); pm1 = fmaxf(pm1, q1);
        h2 hq; hq[0] = (_Float16)q0; hq[1] = (_Float16)q1;
        *(h2*)&p1buf[slot][k * D2 + d0] = hq;
      }
      h2 pl; pl[0] = (_Float16)pm0; pl[1] = (_Float16)pm1;
      *(h2*)&plocalS[slot][d0] = pl;
    }
    asm volatile("s_wait_dscnt 0" ::: "memory");

    /* ---- stage B: p1 = p0 @ mlp2^T + b  ([16,64]x[64,64], WMMA f16) ---- */
    {
      const v16h a0 = loadA(p1buf[slot], D2, 0,  lo, hi);
      const v16h a1 = loadA(p1buf[slot], D2, 32, lo, hi);
#pragma unroll
      for (int t = 0; t < 4; ++t) {
        v8f acc = {};
        acc = wmma16(a0, loadB(m2h, D2, t, 0, lo, hi), acc);
        acc = wmma16(a1, loadB(m2h, D2, t, 1, lo, hi), acc);
        const float bias = m2bS[t * 16 + lo];
#pragma unroll
        for (int v = 0; v < 8; ++v)
          p1buf[slot][(v + 8 * hi) * D2 + t * 16 + lo] =
              (_Float16)(acc[v] + bias);
      }
    }
    asm volatile("s_wait_dscnt 0" ::: "memory");

    /* ---- stage C: h = gelu([p1|p_local] @ w1^T + b1) ([16,128]x[128,128]) ---- */
    {
      const v16h a0 = loadA(p1buf[slot], D2, 0,  lo, hi);
      const v16h a1 = loadA(p1buf[slot], D2, 32, lo, hi);
      const v16h a2 = loadAb(plocalS[slot], 0,  hi); /* broadcast rows */
      const v16h a3 = loadAb(plocalS[slot], 32, hi);
#pragma unroll
      for (int t = 0; t < 8; ++t) {
        v8f acc = {};
        acc = wmma16(a0, loadB(w1h, DIM, t, 0, lo, hi), acc);
        acc = wmma16(a1, loadB(w1h, DIM, t, 1, lo, hi), acc);
        acc = wmma16(a2, loadB(w1h, DIM, t, 2, lo, hi), acc);
        acc = wmma16(a3, loadB(w1h, DIM, t, 3, lo, hi), acc);
        const float bias = b1S[t * 16 + lo];
#pragma unroll
        for (int v = 0; v < 8; ++v) {
          const float xg = acc[v] + bias;
          const float g = 0.5f * xg * (1.0f + erff(xg * 0.70710678118654752f));
          hbuf[slot][(v + 8 * hi) * DIM + t * 16 + lo] = (_Float16)g;
        }
      }
    }
    asm volatile("s_wait_dscnt 0" ::: "memory");

    /* ---- stage D+E: pe = h @ w2^T + b2; fuse xf gather, +, max over K ---- */
    {
      const v16h a0 = loadA(hbuf[slot], DIM, 0,  lo, hi);
      const v16h a1 = loadA(hbuf[slot], DIM, 32, lo, hi);
      const v16h a2 = loadA(hbuf[slot], DIM, 64, lo, hi);
      const v16h a3 = loadA(hbuf[slot], DIM, 96, lo, hi);
      const float* xfb = xf + (size_t)bOff * DIM;
      const float* ctr = xf + (size_t)(bOff + n) * DIM;
      float* orow = out + (size_t)p * DIM;
#pragma unroll
      for (int t = 0; t < 8; ++t) {
        v8f acc = {};
        acc = wmma16(a0, loadB(w2h, DIM, t, 0, lo, hi), acc);
        acc = wmma16(a1, loadB(w2h, DIM, t, 1, lo, hi), acc);
        acc = wmma16(a2, loadB(w2h, DIM, t, 2, lo, hi), acc);
        acc = wmma16(a3, loadB(w2h, DIM, t, 3, lo, hi), acc);
        const int c = t * 16 + lo;
        const float bias = b2S[c];
        float m = -3.0e38f;
#pragma unroll
        for (int v = 0; v < 8; ++v) {
          const int krow = v + 8 * hi;
          const int nbr = nbrS[slot][krow];
          const float xv = xfb[(size_t)nbr * DIM + c]; /* lanes 0-15: 64B seg */
          m = fmaxf(m, xv + acc[v] + bias);
        }
        m = fmaxf(m, __shfl_xor(m, 16, 32)); /* combine row-halves */
        if (lane < 16) {
          const float o = m - ctr[c];
          orow[c] = o;
          atomicAdd(&sSum[c], o);
          atomicAdd(&sSumSq[c], o * o);
        }
      }
    }
  }

  __syncthreads();
  if (tid < DIM)            atomicAdd(&gacc[tid], sSum[tid]);
  else if (tid < 2 * DIM)   atomicAdd(&gacc[tid], sSumSq[tid - DIM]);
}

/* ---------------- kernel 3: BatchNorm normalize in place ---------------- */

__global__ void k_bn(float* __restrict__ out, const float* __restrict__ gacc,
                     const float* __restrict__ gamma,
                     const float* __restrict__ beta) {
  const int idx = blockIdx.x * 256 + threadIdx.x;
  const int c = idx & (DIM - 1);
  const float invM = 1.0f / (float)P_TOTAL;
  const float mean = gacc[c] * invM;
  const float var  = gacc[DIM + c] * invM - mean * mean;
  const float sc   = rsqrtf(var + EPSV) * gamma[c];
  out[idx] = (out[idx] - mean) * sc + beta[c];
}

/* ---------------- host launcher ---------------- */

extern "C" void kernel_launch(void* const* d_in, const int* in_sizes, int n_in,
                              void* d_out, int out_size, void* d_ws,
                              size_t ws_size, hipStream_t stream) {
  (void)in_sizes; (void)n_in; (void)out_size; (void)ws_size;
  const float* xyz    = (const float*)d_in[0];
  const float* x      = (const float*)d_in[1];
  const int*   knn    = (const int*)d_in[2];
  const float* proj_w = (const float*)d_in[3];
  const float* m1w    = (const float*)d_in[4];
  const float* m1b    = (const float*)d_in[5];
  const float* m2w    = (const float*)d_in[6];
  const float* m2b    = (const float*)d_in[7];
  const float* w1     = (const float*)d_in[8];
  const float* b1     = (const float*)d_in[9];
  const float* w2     = (const float*)d_in[10];
  const float* b2     = (const float*)d_in[11];
  const float* gamma  = (const float*)d_in[12];
  const float* beta   = (const float*)d_in[13];
  float* out = (float*)d_out;

  float* xf   = (float*)d_ws;                       /* 65536*128 f32 = 32 MB */
  float* gacc = xf + (size_t)P_TOTAL * DIM;         /* 256 f32: sum|sumsq   */

  k_zero<<<1, 256, 0, stream>>>(gacc);
  k_proj<<<P_TOTAL / (16 * 8), 256, 0, stream>>>(x, proj_w, xf); /* 512 WGs */
  k_lfa<<<P_TOTAL / (8 * 8), 256, 0, stream>>>(xyz, xf, knn, m1w, m1b, m2w,
                                               m2b, w1, b1, w2, b2, out,
                                               gacc);            /* 1024 WGs */
  k_bn<<<(P_TOTAL * DIM) / 256, 256, 0, stream>>>(out, gacc, gamma, beta);
}
